// SelfAttention_45079976739211
// MI455X (gfx1250) — compile-verified, matrix operand order checked by reading, not run on hardware
//
#include <hip/hip_runtime.h>
#include <hip/hip_bf16.h>

typedef __bf16 bf16_t;
typedef __attribute__((ext_vector_type(8)))  __bf16 v8bf;
typedef __attribute__((ext_vector_type(16))) __bf16 v16bf;
typedef __attribute__((ext_vector_type(8)))  float  v8f;

#define NB   8
#define TQ   512
#define TK   1024
#define DQ   1024
#define DKV  1024
#define DATT 1024
#define NH   16
#define DH   64

// ---------------------------------------------------------------------------
// Fragment helpers.
// 16-bit A/B fragment for v_wmma_*_16x16x32: lane%16 selects the row (A) or
// column (B); lanes<16 hold K = {kb..kb+7, kb+16..kb+23}, lanes>=16 hold
// K = {kb+8..kb+15, kb+24..kb+31}. Two 16-byte contiguous loads per lane.
// ---------------------------------------------------------------------------
__device__ __forceinline__ v16bf load_ab_frag(const bf16_t* __restrict__ row,
                                              int kbase, int lane) {
  const int klo = kbase + ((lane & 16) ? 8 : 0);
  v8bf lo = *(const v8bf*)(row + klo);
  v8bf hi = *(const v8bf*)(row + klo + 16);
  v16bf r;
#pragma unroll
  for (int i = 0; i < 8; ++i) { r[i] = lo[i]; r[i + 8] = hi[i]; }
  return r;
}

__device__ __forceinline__ v16bf load_ab_frag_lds(const bf16_t* row, int lane) {
  const int klo = (lane & 16) ? 8 : 0;
  v8bf lo = *(const v8bf*)(row + klo);
  v8bf hi = *(const v8bf*)(row + klo + 16);
  v16bf r;
#pragma unroll
  for (int i = 0; i < 8; ++i) { r[i] = lo[i]; r[i + 8] = hi[i]; }
  return r;
}

__device__ __forceinline__ v8f wmma_bf16(v16bf a, v16bf b, v8f c) {
  return __builtin_amdgcn_wmma_f32_16x16x32_bf16(
      /*neg_a=*/false, a, /*neg_b=*/false, b,
      /*c_mod=*/(short)0, c, /*reuse_a=*/false, /*reuse_b=*/false);
}

// ---------------------------------------------------------------------------
// f32 -> bf16 flat convert (weights)
// ---------------------------------------------------------------------------
__global__ void k_cvt(const float* __restrict__ in, bf16_t* __restrict__ out,
                      long nElem) {
  long i = (long)blockIdx.x * blockDim.x + threadIdx.x;
  if (i < nElem) out[i] = (bf16_t)in[i];
}

// [N][C][T] f32 -> [N][T][C] bf16 (write-coalesced)
__global__ void k_transpose_cvt(const float* __restrict__ in,
                                bf16_t* __restrict__ out, int C, int T) {
  long idx = (long)blockIdx.x * blockDim.x + threadIdx.x;
  long total = (long)C * T;
  if (idx >= total) return;
  int t = (int)(idx / C);
  int c = (int)(idx % C);
  long b = blockIdx.y;
  out[(b * T + t) * C + c] = (bf16_t)in[(b * C + c) * T + t];
}

// ---------------------------------------------------------------------------
// Shared GEMM core: one wave computes a 32(M) x 64(N) tile of
//   C[m][n] = sum_k A[m][k] * W[n][k]
// A: [M][K] bf16 row-major (per batch), W: [N][K] bf16 row-major.
// 2 A-fragments x 4 B-fragments -> 8 WMMA per 12 b128 loads per K-chunk.
// Accumulators: accp[pair][j], pair selects rows m0+pair*16.
// ---------------------------------------------------------------------------
__device__ __forceinline__ void gemm_core_32x64(const bf16_t* __restrict__ Ab,
                                                const bf16_t* __restrict__ W,
                                                int m0, int n0, int K, int lane,
                                                v8f acc[2][4]) {
  const bf16_t* arow0 = Ab + (long)(m0 + (lane & 15)) * K;
  const bf16_t* arow1 = Ab + (long)(m0 + 16 + (lane & 15)) * K;
  const bf16_t* wrow[4];
#pragma unroll
  for (int j = 0; j < 4; ++j)
    wrow[j] = W + (long)(n0 + j * 16 + (lane & 15)) * K;

  for (int kb = 0; kb < K; kb += 32) {
    v16bf af0 = load_ab_frag(arow0, kb, lane);
    v16bf af1 = load_ab_frag(arow1, kb, lane);
    v16bf bfr[4];
#pragma unroll
    for (int j = 0; j < 4; ++j) bfr[j] = load_ab_frag(wrow[j], kb, lane);
#pragma unroll
    for (int j = 0; j < 4; ++j) {
      acc[0][j] = wmma_bf16(af0, bfr[j], acc[0][j]);
      acc[1][j] = wmma_bf16(af1, bfr[j], acc[1][j]);
    }
  }
}

// ---------------------------------------------------------------------------
// Projection GEMM: out[t][d] = sum_c A[t][c]*W[d][c] + bias[d], bf16 out.
// ---------------------------------------------------------------------------
__global__ void k_gemm_proj(const bf16_t* __restrict__ A,
                            const bf16_t* __restrict__ W,
                            const float* __restrict__ bias,
                            bf16_t* __restrict__ out, int M, int K, int N) {
  const int lane = threadIdx.x;
  const int n0 = blockIdx.x * 64;
  const int m0 = blockIdx.y * 32;
  const long b = blockIdx.z;
  v8f acc[2][4] = {};
  gemm_core_32x64(A + b * M * (long)K, W, m0, n0, K, lane, acc);

  const int col = lane & 15;
  const int rbase = (lane & 16) ? 8 : 0;
  bf16_t* ob = out + b * M * (long)N;
#pragma unroll
  for (int pr = 0; pr < 2; ++pr)
#pragma unroll
    for (int j = 0; j < 4; ++j) {
      float bv = bias[n0 + j * 16 + col];
#pragma unroll
      for (int r = 0; r < 8; ++r)
        ob[(long)(m0 + pr * 16 + rbase + r) * N + n0 + j * 16 + col] =
            (bf16_t)(acc[pr][j][r] + bv);
    }
}

// ---------------------------------------------------------------------------
// KV projection: row-major kvf plus transposed scatter of the V half.
// kvf col e = h*128 + c: c<64 -> K half (QK^T reads rows), c>=64 -> V half,
// stored as vT[b][h*64+d][t] so PV B-fragments are K-contiguous.
// ---------------------------------------------------------------------------
__global__ void k_gemm_kvproj(const bf16_t* __restrict__ A,
                              const bf16_t* __restrict__ W,
                              const float* __restrict__ bias,
                              bf16_t* __restrict__ kvf,
                              bf16_t* __restrict__ vT, int M, int K, int N) {
  const int lane = threadIdx.x;
  const int n0 = blockIdx.x * 64;
  const int m0 = blockIdx.y * 32;
  const long b = blockIdx.z;
  v8f acc[2][4] = {};
  gemm_core_32x64(A + b * M * (long)K, W, m0, n0, K, lane, acc);

  const int col = lane & 15;
  const int rbase = (lane & 16) ? 8 : 0;
  bf16_t* kb_ = kvf + b * M * (long)N;
  bf16_t* vb_ = vT + b * (long)DATT * TK;
#pragma unroll
  for (int pr = 0; pr < 2; ++pr)
#pragma unroll
    for (int j = 0; j < 4; ++j) {
      const int e = n0 + j * 16 + col;
      float bv = bias[e];
#pragma unroll
      for (int r = 0; r < 8; ++r) {
        const int row = m0 + pr * 16 + rbase + r;
        float v = acc[pr][j][r] + bv;
        kb_[(long)row * N + e] = (bf16_t)v;
        if (e & 64) {  // V half of this head's 128-wide slice
          const int h = e >> 7, d = e & 63;
          vb_[(long)(h * DH + d) * TK + row] = (bf16_t)v;
        }
      }
    }
}

// ---------------------------------------------------------------------------
// Fused flash attention: one wave per (batch, head, 16 q rows).
// Streams keys in chunks of 32; online softmax in f32; P restaged via LDS
// from C-fragment layout into A-fragment layout for the PV matmul.
// ---------------------------------------------------------------------------
__global__ void k_attention(const bf16_t* __restrict__ qf,   // [b][TQ][DATT]
                            const bf16_t* __restrict__ kvf,  // [b][TK][2*DATT]
                            const bf16_t* __restrict__ vT,   // [b][DATT][TK]
                            const int* __restrict__ kv_len,
                            bf16_t* __restrict__ ctx) {      // [b][TQ][DATT]
  const int lane = threadIdx.x;
  const int m0 = blockIdx.x * 16;
  const int h  = blockIdx.y;
  const long b = blockIdx.z;
  const int klen = kv_len[b];
  const int col = lane & 15;
  const int rbase = (lane & 16) ? 8 : 0;

  __shared__ bf16_t pbuf[16 * 32];

  // Q tile held in registers: 16 rows x 64 (K split into two 32-wide frags)
  const bf16_t* qrow = qf + (b * TQ + m0 + col) * (long)DATT + h * DH;
  v16bf qa0 = load_ab_frag(qrow, 0, lane);
  v16bf qa1 = load_ab_frag(qrow, 32, lane);

  v8f acc[4] = {};
  float mrun[8], lrun[8];
#pragma unroll
  for (int r = 0; r < 8; ++r) { mrun[r] = -__builtin_inff(); lrun[r] = 0.f; }
  const float scale = 1.0f / 64.0f;  // H / D

  for (int kb = 0; kb < TK; kb += 32) {
    // scores: two 16-wide k sub-tiles, K = dh = 64 -> 2 WMMA each
    v8f s[2];
#pragma unroll
    for (int st = 0; st < 2; ++st) {
      const int krow = kb + st * 16 + col;
      const bf16_t* kp = kvf + (b * TK + krow) * (long)(2 * DATT) + h * 128;
      v8f z = {};
      z = wmma_bf16(qa0, load_ab_frag(kp, 0, lane), z);
      z = wmma_bf16(qa1, load_ab_frag(kp, 32, lane), z);
      s[st] = z;
    }
    // scale + kv_len mask (per-lane: one k column for all 8 rows)
#pragma unroll
    for (int st = 0; st < 2; ++st) {
      const bool valid = (kb + st * 16 + col) < klen;
#pragma unroll
      for (int r = 0; r < 8; ++r)
        s[st][r] = valid ? s[st][r] * scale : -1.0e10f;
    }
    // online softmax: row stats across the 16 lanes of each half
    float alpha[8];
#pragma unroll
    for (int r = 0; r < 8; ++r) {
      float m = fmaxf(s[0][r], s[1][r]);
      m = fmaxf(m, __shfl_xor(m, 1, 16));
      m = fmaxf(m, __shfl_xor(m, 2, 16));
      m = fmaxf(m, __shfl_xor(m, 4, 16));
      m = fmaxf(m, __shfl_xor(m, 8, 16));
      const float mnew = fmaxf(mrun[r], m);
      alpha[r] = __expf(mrun[r] - mnew);
      mrun[r] = mnew;
      const float p0 = __expf(s[0][r] - mnew);
      const float p1 = __expf(s[1][r] - mnew);
      s[0][r] = p0; s[1][r] = p1;
      float rs = p0 + p1;
      rs += __shfl_xor(rs, 1, 16);
      rs += __shfl_xor(rs, 2, 16);
      rs += __shfl_xor(rs, 4, 16);
      rs += __shfl_xor(rs, 8, 16);
      lrun[r] = lrun[r] * alpha[r] + rs;
    }
    // restage P (C-fragment -> LDS 16x32 row-major -> A-fragment)
    __syncthreads();
#pragma unroll
    for (int r = 0; r < 8; ++r) {
      pbuf[(rbase + r) * 32 + col]      = (bf16_t)s[0][r];
      pbuf[(rbase + r) * 32 + 16 + col] = (bf16_t)s[1][r];
    }
    __syncthreads();
    v16bf pa = load_ab_frag_lds(pbuf + (lane & 15) * 32, lane);
    // ctx accumulate: rescale then 4 WMMA over the 64-wide d dimension
#pragma unroll
    for (int j = 0; j < 4; ++j) {
#pragma unroll
      for (int r = 0; r < 8; ++r) acc[j][r] *= alpha[r];
      const bf16_t* vrow = vT + (b * DATT + h * DH + j * 16 + col) * (long)TK;
      acc[j] = wmma_bf16(pa, load_ab_frag(vrow, kb, lane), acc[j]);
    }
  }
  // finalize: 1/l and store (klen >= 1 guarantees lrun > 0)
#pragma unroll
  for (int j = 0; j < 4; ++j)
#pragma unroll
    for (int r = 0; r < 8; ++r)
      ctx[(b * TQ + m0 + rbase + r) * (long)DATT + h * DH + j * 16 + col] =
          (bf16_t)(acc[j][r] / lrun[r]);
}

// ---------------------------------------------------------------------------
// FC: out[b][o][t] = sum_e ctx[b][t][e] * Wfc[o][e] + bfc[o]  (f32, transposed)
// ---------------------------------------------------------------------------
__global__ void k_gemm_fc(const bf16_t* __restrict__ A,
                          const bf16_t* __restrict__ W,
                          const float* __restrict__ bias,
                          float* __restrict__ out, int M, int K, int N) {
  const int lane = threadIdx.x;
  const int n0 = blockIdx.x * 64;
  const int m0 = blockIdx.y * 32;
  const long b = blockIdx.z;
  v8f acc[2][4] = {};
  gemm_core_32x64(A + b * M * (long)K, W, m0, n0, K, lane, acc);

  const int col = lane & 15;
  const int rbase = (lane & 16) ? 8 : 0;
  float* ob = out + b * (long)N * M;
#pragma unroll
  for (int pr = 0; pr < 2; ++pr)
#pragma unroll
    for (int j = 0; j < 4; ++j) {
      const int o = n0 + j * 16 + col;
      float bv = bias[o];
#pragma unroll
      for (int r = 0; r < 8; ++r)
        ob[(long)o * M + m0 + pr * 16 + rbase + r] = acc[pr][j][r] + bv;
    }
}

// ---------------------------------------------------------------------------
extern "C" void kernel_launch(void* const* d_in, const int* in_sizes, int n_in,
                              void* d_out, int out_size, void* d_ws, size_t ws_size,
                              hipStream_t stream) {
  const float* q   = (const float*)d_in[0];
  const float* kv  = (const float*)d_in[1];
  const int*   kvl = (const int*)d_in[2];
  const float* Wq  = (const float*)d_in[3];
  const float* bq  = (const float*)d_in[4];
  const float* Wkv = (const float*)d_in[5];
  const float* bkv = (const float*)d_in[6];
  const float* Wfc = (const float*)d_in[7];
  const float* bfc = (const float*)d_in[8];
  float* out = (float*)d_out;

  // workspace carve-up (bf16), ~96 MiB total
  bf16_t* p = (bf16_t*)d_ws;
  bf16_t* qT   = p; p += (long)NB * TQ * DQ;        // q  transposed [b][t][c]
  bf16_t* kvT  = p; p += (long)NB * TK * DKV;       // kv transposed [b][t][c]
  bf16_t* wqb  = p; p += (long)DATT * DQ;
  bf16_t* wkvb = p; p += (long)2 * DATT * DKV;
  bf16_t* wfcb = p; p += (long)DATT * DATT;
  bf16_t* qfb  = p; p += (long)NB * TQ * DATT;      // Q projection
  bf16_t* kvfb = p; p += (long)NB * TK * 2 * DATT;  // KV projection (row-major)
  bf16_t* vTb  = p; p += (long)NB * DATT * TK;      // V transposed [b][h*64+d][t]
  bf16_t* ctxb = p; p += (long)NB * TQ * DATT;      // attention output

  // 1) precision conversion (one-time, L2-resident afterwards)
  k_cvt<<<dim3((DATT * DQ + 255) / 256), 256, 0, stream>>>(Wq, wqb, (long)DATT * DQ);
  k_cvt<<<dim3((2 * DATT * DKV + 255) / 256), 256, 0, stream>>>(Wkv, wkvb, (long)2 * DATT * DKV);
  k_cvt<<<dim3((DATT * DATT + 255) / 256), 256, 0, stream>>>(Wfc, wfcb, (long)DATT * DATT);
  k_transpose_cvt<<<dim3((DQ * TQ + 255) / 256, NB), 256, 0, stream>>>(q, qT, DQ, TQ);
  k_transpose_cvt<<<dim3((DKV * TK + 255) / 256, NB), 256, 0, stream>>>(kv, kvT, DKV, TK);

  // 2) projections (WMMA bf16, f32 accumulate), 32x64 tiles per wave
  k_gemm_proj<<<dim3(DATT / 64, TQ / 32, NB), 32, 0, stream>>>(qT, wqb, bq, qfb, TQ, DQ, DATT);
  k_gemm_kvproj<<<dim3(2 * DATT / 64, TK / 32, NB), 32, 0, stream>>>(kvT, wkvb, bkv, kvfb, vTb,
                                                                     TK, DKV, 2 * DATT);
  // 3) fused flash attention
  k_attention<<<dim3(TQ / 16, NH, NB), 32, 0, stream>>>(qfb, kvfb, vTb, kvl, ctxb);

  // 4) output projection (+ transposed f32 store)
  k_gemm_fc<<<dim3(DATT / 64, TQ / 32, NB), 32, 0, stream>>>(ctxb, wfcb, bfc, out, TQ, DATT, DATT);
}
